// ModulatedMLPHead_70557722739271
// MI455X (gfx1250) — matrix operand order repaired and checked by reading, not compiled, not run on hardware
//
#include <hip/hip_runtime.h>
#include <hip/hip_bf16.h>
#include <math.h>

// ---------------------------------------------------------------------------
// ModulatedMLPHead fused CDNA5 implementation.
//   B=128, T=2048, D_MODEL=512, HIDDEN=256, LAYERS=4
//   Hot path: 3x (B*T,256)@(256,256) GEMMs -> v_wmma_f32_16x16x32_f16
// ---------------------------------------------------------------------------

typedef __attribute__((ext_vector_type(16))) _Float16 v16h;
typedef __attribute__((ext_vector_type(8)))  _Float16 v8h;
typedef __attribute__((ext_vector_type(8)))  float    v8f;

union Frag {
    v16h f;
    v8h  h[2];
};

#define BSZ   128
#define TSZ   2048
#define DM    512
#define HID   256
#define TM    128      // rows per workgroup in the main kernel
#define XS    264      // padded LDS row stride in halves (264*2B = 528B, 16B-aligned)
#define YSZ   (BSZ * TSZ * 6)

// Workspace layout (bytes):
//   [0)                 As16   : 3*256*256 f16 = 393216 B
//   [393216)            Aout16 : 16*256   f16 =   8192 B
//   [401408)            hs     : 4*128*256 f32 = 524288 B
#define WS_AS16   0
#define WS_AOUT16 393216
#define WS_HS     401408

// ---------------------------------------------------------------------------
// Kernel 0: convert weights to f16 (As_w -> As16, A_out_w -> padded 16x256)
// ---------------------------------------------------------------------------
__global__ void convert_kernel(const float* __restrict__ As_w,
                               const float* __restrict__ A_out_w,
                               _Float16* __restrict__ As16,
                               _Float16* __restrict__ Aout16)
{
    int i = blockIdx.x * blockDim.x + threadIdx.x;
    if (i < 3 * HID * HID)
        As16[i] = (_Float16)As_w[i];
    if (i < 16 * HID) {
        int r = i >> 8, c = i & 255;
        Aout16[i] = (r < 6) ? (_Float16)A_out_w[r * HID + c] : (_Float16)0.0f;
    }
}

// ---------------------------------------------------------------------------
// Kernel 1: per-batch modulation MLP -> hs[4][128][256], and sigmoid head f.
// One block per batch row b; 256 threads (one per hidden unit).
// ---------------------------------------------------------------------------
__global__ __launch_bounds__(256)
void mod_kernel(const float* __restrict__ Pj,
                const float* __restrict__ w0_w, const float* __restrict__ w0_b,
                const float* __restrict__ ws_w, const float* __restrict__ ws_b,
                const float* __restrict__ c1_w, const float* __restrict__ c1_b,
                const float* __restrict__ c2_w, const float* __restrict__ c2_b,
                float* __restrict__ hs, float* __restrict__ fout)
{
    __shared__ float pjs[DM];
    __shared__ float hb[HID];
    __shared__ float red[HID];

    const int b   = blockIdx.x;
    const int tid = threadIdx.x;

    pjs[tid]       = Pj[b * DM + tid];
    pjs[tid + 256] = Pj[b * DM + 256 + tid];
    __syncthreads();

    // h0 = relu(Pj @ w0_w.T + w0_b)
    {
        float a = w0_b[tid];
        const float* wr = w0_w + tid * DM;
        for (int k = 0; k < DM; ++k) a = fmaf(wr[k], pjs[k], a);
        a = fmaxf(a, 0.0f);
        hb[tid] = a;
        hs[(0 * BSZ + b) * HID + tid] = a;
    }
    __syncthreads();

    // h_{i+1} = relu(concat(h, Pj) @ ws_w[i].T + ws_b[i])
    for (int i = 0; i < 3; ++i) {
        float acc = ws_b[i * HID + tid];
        const float* r = ws_w + (size_t)(i * HID + tid) * (HID + DM);
        for (int k = 0; k < HID; ++k) acc = fmaf(r[k], hb[k], acc);
        for (int k = 0; k < DM;  ++k) acc = fmaf(r[HID + k], pjs[k], acc);
        acc = fmaxf(acc, 0.0f);
        __syncthreads();
        hb[tid] = acc;
        hs[((i + 1) * BSZ + b) * HID + tid] = acc;
        __syncthreads();
    }

    // f = sigmoid(relu(Pj @ c1_w.T + c1_b) @ c2_w.T + c2_b)
    float p = 0.0f;
    for (int oo = 0; oo < 2; ++oo) {
        const int o = tid + oo * 256;
        float u = c1_b[o];
        const float* cr = c1_w + (size_t)o * DM;
        for (int k = 0; k < DM; ++k) u = fmaf(cr[k], pjs[k], u);
        u = fmaxf(u, 0.0f);
        p = fmaf(c2_w[o], u, p);
    }
    red[tid] = p;
    __syncthreads();
    for (int st = 128; st > 0; st >>= 1) {
        if (tid < st) red[tid] += red[tid + st];
        __syncthreads();
    }
    if (tid == 0)
        fout[b] = 1.0f / (1.0f + expf(-(red[0] + c2_b[0])));
}

// ---------------------------------------------------------------------------
// Kernel 2: fused main MLP. One workgroup = 128 rows of one batch b.
// 8 waves in a 2x4 grid; each wave owns a 64x64 f32 accumulator block.
// x tile resides in LDS as f16 (128 x 264-stride).
// ---------------------------------------------------------------------------
__global__ __launch_bounds__(256, 1)
void main_kernel(const float* __restrict__ s,
                 const float* __restrict__ A0_w, const float* __restrict__ A0_b,
                 const float* __restrict__ As_b,         // (3,256)
                 const float* __restrict__ A_out_b,      // (6)
                 const _Float16* __restrict__ As16,      // (3,256,256) N x K
                 const _Float16* __restrict__ Aout16,    // (16,256)    N x K padded
                 const float* __restrict__ hs,           // (4,128,256)
                 float* __restrict__ y)                  // (128,2048,6)
{
    extern __shared__ __align__(16) char dynsmem[];
    _Float16* xbuf = (_Float16*)dynsmem;                 // [TM][XS] f16
    __shared__ float sbuf[TM];

    const int b   = blockIdx.y;
    const int t0  = blockIdx.x * TM;
    const int tid = threadIdx.x;

    // Stage 0: stage the s tile (one scalar per row).
    if (tid < TM) sbuf[tid] = s[b * TSZ + t0 + tid];
    __syncthreads();

    // Stage 1: x0[m,n] = hs0[b,n] * relu(s[m]*A0_w[n] + A0_b[n])  -> LDS f16
    {
        const int   n   = tid;                            // 0..255
        const float a0w = A0_w[n];
        const float a0b = A0_b[n];
        const float h0  = hs[(0 * BSZ + b) * HID + n];
        for (int m = 0; m < TM; ++m) {
            float v = fmaxf(fmaf(sbuf[m], a0w, a0b), 0.0f) * h0;
            xbuf[m * XS + n] = (_Float16)v;
        }
    }
    __syncthreads();

    const int lane = tid & 31;
    const int w    = tid >> 5;            // wave id 0..7
    const int wm   = (w >> 2) * 64;       // wave row base: 0 or 64
    const int wn   = (w & 3) * 64;        // wave col base: 0,64,128,192
    const int lr   = lane & 15;           // row (A) / col (B,C) within tile
    const int hi   = lane >> 4;           // lane-half select
    const int lkA  = hi << 3;             //  A frag K offset: 0 or 8
    const int lkB  = hi << 3;             //  B frag K offset: 0 or 8

    // Stage 2: three FiLM-modulated 256x256 layers, x resident in LDS.
    for (int layer = 0; layer < 3; ++layer) {
        const _Float16* Wl = As16 + (size_t)layer * HID * HID;

        v8f acc[4][4];
        const v8f zero = {0.f, 0.f, 0.f, 0.f, 0.f, 0.f, 0.f, 0.f};
        #pragma unroll
        for (int mt = 0; mt < 4; ++mt)
            #pragma unroll
            for (int nt = 0; nt < 4; ++nt)
                acc[mt][nt] = zero;

        #pragma unroll
        for (int nt = 0; nt < 4; ++nt) {
            const _Float16* wrow = Wl + (size_t)(wn + nt * 16 + lr) * HID;
            #pragma unroll
            for (int kt = 0; kt < 8; ++kt) {
                Frag bf;                                   // W row, K {0..7,16..23}/{8..15,24..31}
                const _Float16* bp = wrow + kt * 32 + lkB;
                bf.h[0] = *(const v8h*)(bp);
                bf.h[1] = *(const v8h*)(bp + 16);
                #pragma unroll
                for (int mt = 0; mt < 4; ++mt) {
                    Frag af;                               // x row, K {0..7,16..23}/{8..15,24..31}
                    const _Float16* ap = xbuf + (wm + mt * 16 + lr) * XS + kt * 32 + lkA;
                    af.h[0] = *(const v8h*)(ap);
                    af.h[1] = *(const v8h*)(ap + 16);
                    acc[mt][nt] = __builtin_amdgcn_wmma_f32_16x16x32_f16(
                        false, af.f, false, bf.f, (short)0, acc[mt][nt], false, false);
                }
            }
        }
        __syncthreads();   // all waves done reading xbuf for this layer

        // Epilogue: bias + relu + FiLM, write f16 back to LDS.
        #pragma unroll
        for (int nt = 0; nt < 4; ++nt) {
            const int   n    = wn + nt * 16 + lr;
            const float bias = As_b[layer * HID + n];
            const float hmod = hs[((layer + 1) * BSZ + b) * HID + n];
            #pragma unroll
            for (int mt = 0; mt < 4; ++mt) {
                const int mbase = wm + mt * 16 + 8 * hi;   // C layout: m = vgpr + 8*(lane>=16)
                #pragma unroll
                for (int v2 = 0; v2 < 8; ++v2) {
                    float val = fmaxf(acc[mt][nt][v2] + bias, 0.0f) * hmod;
                    xbuf[(mbase + v2) * XS + n] = (_Float16)val;
                }
            }
        }
        __syncthreads();
    }

    // Stage 3: output layer y = x @ A_out.T + b  (N padded 6 -> 16; one tile/wave)
    {
        const int row0 = 16 * w;                            // 8 waves x 16 rows = 128
        v8f acc = {0.f, 0.f, 0.f, 0.f, 0.f, 0.f, 0.f, 0.f};
        #pragma unroll
        for (int kt = 0; kt < 8; ++kt) {
            Frag af, bf;
            const _Float16* ap = xbuf + (row0 + lr) * XS + kt * 32 + lkA;
            af.h[0] = *(const v8h*)(ap);
            af.h[1] = *(const v8h*)(ap + 16);
            const _Float16* bp = Aout16 + (size_t)lr * HID + kt * 32 + lkB;
            bf.h[0] = *(const v8h*)(bp);
            bf.h[1] = *(const v8h*)(bp + 16);
            acc = __builtin_amdgcn_wmma_f32_16x16x32_f16(
                false, af.f, false, bf.f, (short)0, acc, false, false);
        }
        const int n = lr;
        if (n < 6) {
            const float bias = A_out_b[n];
            const int   mb   = row0 + 8 * hi;
            #pragma unroll
            for (int v2 = 0; v2 < 8; ++v2) {
                const size_t row = (size_t)b * TSZ + t0 + mb + v2;
                y[row * 6 + n] = acc[v2] + bias;
            }
        }
    }
}

// ---------------------------------------------------------------------------
extern "C" void kernel_launch(void* const* d_in, const int* in_sizes, int n_in,
                              void* d_out, int out_size, void* d_ws, size_t ws_size,
                              hipStream_t stream)
{
    const float* Pj      = (const float*)d_in[0];
    const float* s       = (const float*)d_in[1];
    const float* A0_w    = (const float*)d_in[2];
    const float* A0_b    = (const float*)d_in[3];
    const float* As_w    = (const float*)d_in[4];
    const float* As_b    = (const float*)d_in[5];
    const float* A_out_w = (const float*)d_in[6];
    const float* A_out_b = (const float*)d_in[7];
    const float* w0_w    = (const float*)d_in[8];
    const float* w0_b    = (const float*)d_in[9];
    const float* ws_w    = (const float*)d_in[10];
    const float* ws_b    = (const float*)d_in[11];
    const float* c1_w    = (const float*)d_in[12];
    const float* c1_b    = (const float*)d_in[13];
    const float* c2_w    = (const float*)d_in[14];
    const float* c2_b    = (const float*)d_in[15];

    float* out = (float*)d_out;
    char*  ws  = (char*)d_ws;
    _Float16* As16   = (_Float16*)(ws + WS_AS16);
    _Float16* Aout16 = (_Float16*)(ws + WS_AOUT16);
    float*    hsbuf  = (float*)   (ws + WS_HS);

    // 0: weight conversion (covers 3*256*256 elements)
    convert_kernel<<<(3 * HID * HID + 255) / 256, 256, 0, stream>>>(
        As_w, A_out_w, As16, Aout16);

    // 1: modulation MLP + sigmoid head
    mod_kernel<<<BSZ, 256, 0, stream>>>(
        Pj, w0_w, w0_b, ws_w, ws_b, c1_w, c1_b, c2_w, c2_b,
        hsbuf, out + (size_t)YSZ);

    // 2: fused main MLP (WMMA). Dynamic LDS: 128 x 264 f16 = 67584 B.
    dim3 grid(TSZ / TM, BSZ);
    main_kernel<<<grid, 256, TM * XS * sizeof(_Float16), stream>>>(
        s, A0_w, A0_b, As_b, A_out_b, As16, Aout16, hsbuf, out);
}
